// LatentDecoderGRU_3530463117658
// MI455X (gfx1250) — compile-verified, matrix operand order checked by reading, not run on hardware
//
#include <hip/hip_runtime.h>
#include <hip/hip_bf16.h>

// ---------------------------------------------------------------------------
// LatentDecoderGRU for MI455X (gfx1250): 2-layer GRU, B=512, L=256, H=1024.
// Recurrent GEMMs on v_wmma_f32_16x16x32_f16, f16 weights L2-resident (18 MB
// << 192 MB L2). Register blocking: each wave owns 32(batch) x 16(hidden)
// (2 M-tiles sharing every weight tile load) -> 0.83 tile-loads per WMMA.
// Hidden state: f32 master + f16 GEMM copy, double-buffered per step.
// 3 kernel launches per timestep, graph-captured by the harness.
// ---------------------------------------------------------------------------

typedef __attribute__((ext_vector_type(16))) _Float16 v16h;
typedef __attribute__((ext_vector_type(8)))  _Float16 v8h;
typedef __attribute__((ext_vector_type(8)))  float    v8f;

#define B_   512
#define L_   256
#define LAT_ 256
#define H_   1024
#define V_   20

union V16U { v16h v; v8h h[2]; };

// Load a 16x32 f16 operand tile (row-major, ld halves per row) in the CDNA5
// WMMA operand register layout (ISA 7.12.2): lane = {half,row}, VGPR0..3
// hold K = half*8 + 0..7, VGPR4..7 hold K = 16 + half*8 + 0..7.
__device__ inline v16h load_tile16x32(const _Float16* __restrict__ base,
                                      int ld, int k0) {
  int lane = threadIdx.x & 31;
  int row  = lane & 15;
  int koff = (lane >> 4) << 3;  // 0 or 8
  const _Float16* p = base + (size_t)row * ld + k0 + koff;
  V16U r;
  r.h[0] = *(const v8h*)(p);        // K = k0 + koff + 0..7
  r.h[1] = *(const v8h*)(p + 16);   // K = k0 + 16 + koff + 0..7
  return r.v;
}

__device__ inline v8f wmma_f16(v16h a, v16h b, v8f c) {
  return __builtin_amdgcn_wmma_f32_16x16x32_f16(
      /*neg_a=*/false, a, /*neg_b=*/false, b,
      /*c_mod=*/(short)0, c, /*reuse_a=*/false, /*reuse_b=*/false);
}

__device__ inline float sigmoidf_(float x) { return 1.0f / (1.0f + __expf(-x)); }

// --------------------------- one-time kernels ------------------------------

__global__ __launch_bounds__(256) void f32_to_f16_k(
    const float* __restrict__ src, _Float16* __restrict__ dst, int n) {
  int i = blockIdx.x * 256 + threadIdx.x;
  if (i < n) dst[i] = (_Float16)src[i];
}

// h0 = (z @ W_l2h^T + b_l2h).reshape(2, B, H); writes f32 + f16 copies.
__global__ __launch_bounds__(256) void init_h_k(
    const float* __restrict__ z, const float* __restrict__ W,
    const float* __restrict__ bvec,
    float* __restrict__ h0f32, _Float16* __restrict__ h0f16,
    float* __restrict__ h1f32, _Float16* __restrict__ h1f16) {
  int idx = blockIdx.x * 256 + threadIdx.x;   // 0 .. B*2H-1
  int b = idx >> 11;
  int g = idx & 2047;
  const float* zr = z + (size_t)b * LAT_;
  const float* wr = W + (size_t)g * LAT_;
  float s = bvec[g];
#pragma unroll 4
  for (int k = 0; k < LAT_; ++k) s += zr[k] * wr[k];
  int j = g & (H_ - 1);
  if ((g >> 10) == 0) {
    h0f32[(size_t)b * H_ + j] = s;
    h0f16[(size_t)b * H_ + j] = (_Float16)s;
  } else {
    h1f32[(size_t)b * H_ + j] = s;
    h1f16[(size_t)b * H_ + j] = (_Float16)s;
  }
}

// ------------------------------ GRU layer 0 --------------------------------
// Per wave: 2 M-tiles (32 batch rows) x 1 J-tile (16 hidden cols), all three
// gates. Weight tiles (r,z,n) loaded once per chunk, used by both M-tiles.
__device__ inline void gru0_update(
    v8f cr, v8f cz, v8f cn, int m0, int j,
    float bihr, float bihz, float bihn, float bhhr, float bhhz, float bhhn,
    const float* __restrict__ hprev32, float* __restrict__ hnew32,
    _Float16* __restrict__ hnew16, const float* __restrict__ Wih,
    const int* __restrict__ tseq, int t) {
  int lane  = threadIdx.x & 31;
  int mbase = (lane >> 4) << 3;
#pragma unroll
  for (int e = 0; e < 8; ++e) {
    int b = m0 + mbase + e;
    float xr = 0.f, xz = 0.f, xn = 0.f;
    if (t > 0) {
      int tok = tseq[(size_t)b * L_ + (t - 1)];
      xr = Wih[(size_t)j * V_ + tok];
      xz = Wih[(size_t)(H_ + j) * V_ + tok];
      xn = Wih[(size_t)(2 * H_ + j) * V_ + tok];
    }
    float hp = hprev32[(size_t)b * H_ + j];
    float r  = sigmoidf_(xr + bihr + cr[e] + bhhr);
    float zg = sigmoidf_(xz + bihz + cz[e] + bhhz);
    float n  = tanhf(xn + bihn + r * (cn[e] + bhhn));
    float hv = (1.0f - zg) * n + zg * hp;
    hnew32[(size_t)b * H_ + j] = hv;
    hnew16[(size_t)b * H_ + j] = (_Float16)hv;
  }
}

__global__ __launch_bounds__(256) void gru0_step_k(
    const _Float16* __restrict__ hprev16, const float* __restrict__ hprev32,
    float* __restrict__ hnew32, _Float16* __restrict__ hnew16,
    const _Float16* __restrict__ Whh,   // f16 [3H, H]
    const float* __restrict__ Wih,      // f32 [3H, V]
    const int* __restrict__ tseq,       // [B, L]
    const float* __restrict__ bih, const float* __restrict__ bhh, int t) {
  int wave = threadIdx.x >> 5;
  int tile = blockIdx.x * 8 + wave;     // 1024 tiles = 16 (B/32) x 64 (H/16)
  int m0 = (tile >> 6) << 5;            // 32-row M group
  int j0 = (tile & 63) << 4;

  v8f cr0 = {}, cz0 = {}, cn0 = {};
  v8f cr1 = {}, cz1 = {}, cn1 = {};
  const _Float16* A0 = hprev16 + (size_t)m0 * H_;
  const _Float16* A1 = hprev16 + (size_t)(m0 + 16) * H_;
  const _Float16* Br = Whh + (size_t)(j0) * H_;
  const _Float16* Bz = Whh + (size_t)(H_ + j0) * H_;
  const _Float16* Bn = Whh + (size_t)(2 * H_ + j0) * H_;
  for (int k0 = 0; k0 < H_; k0 += 32) {
    v16h a0 = load_tile16x32(A0, H_, k0);
    v16h a1 = load_tile16x32(A1, H_, k0);
    v16h br = load_tile16x32(Br, H_, k0);
    v16h bz = load_tile16x32(Bz, H_, k0);
    v16h bn = load_tile16x32(Bn, H_, k0);
    cr0 = wmma_f16(a0, br, cr0);
    cr1 = wmma_f16(a1, br, cr1);
    cz0 = wmma_f16(a0, bz, cz0);
    cz1 = wmma_f16(a1, bz, cz1);
    cn0 = wmma_f16(a0, bn, cn0);
    cn1 = wmma_f16(a1, bn, cn1);
  }

  int j = j0 + ((threadIdx.x & 31) & 15);
  float bihr = bih[j], bihz = bih[H_ + j], bihn = bih[2 * H_ + j];
  float bhhr = bhh[j], bhhz = bhh[H_ + j], bhhn = bhh[2 * H_ + j];
  gru0_update(cr0, cz0, cn0, m0, j, bihr, bihz, bihn, bhhr, bhhz, bhhn,
              hprev32, hnew32, hnew16, Wih, tseq, t);
  gru0_update(cr1, cz1, cn1, m0 + 16, j, bihr, bihz, bihn, bhhr, bhhz, bhhn,
              hprev32, hnew32, hnew16, Wih, tseq, t);
}

// ------------------------------ GRU layer 1 --------------------------------
// Fuses the input GEMM over y0 (layer-0 new state). r,z accumulate hh+ih
// together; n keeps hh and ih split (r gates only the hh part + bhh).
// 2 M-tiles per wave share all 6 weight tiles per chunk.
__device__ inline void gru1_update(
    v8f cr, v8f cz, v8f chn, v8f cin, int m0, int j,
    float bihr, float bihz, float bihn, float bhhr, float bhhz, float bhhn,
    const float* __restrict__ hprev32, float* __restrict__ hnew32,
    _Float16* __restrict__ hnew16) {
  int lane  = threadIdx.x & 31;
  int mbase = (lane >> 4) << 3;
#pragma unroll
  for (int e = 0; e < 8; ++e) {
    int b = m0 + mbase + e;
    float hp = hprev32[(size_t)b * H_ + j];
    float r  = sigmoidf_(cr[e] + bihr + bhhr);
    float zg = sigmoidf_(cz[e] + bihz + bhhz);
    float n  = tanhf((cin[e] + bihn) + r * (chn[e] + bhhn));
    float hv = (1.0f - zg) * n + zg * hp;
    hnew32[(size_t)b * H_ + j] = hv;
    hnew16[(size_t)b * H_ + j] = (_Float16)hv;
  }
}

__global__ __launch_bounds__(256) void gru1_step_k(
    const _Float16* __restrict__ hprev16, const float* __restrict__ hprev32,
    const _Float16* __restrict__ x16,     // y0[t] f16 [B, H]
    float* __restrict__ hnew32, _Float16* __restrict__ hnew16,
    const _Float16* __restrict__ Whh, const _Float16* __restrict__ Wih,
    const float* __restrict__ bih, const float* __restrict__ bhh) {
  int wave = threadIdx.x >> 5;
  int tile = blockIdx.x * 8 + wave;     // 1024 tiles = 16 x 64
  int m0 = (tile >> 6) << 5;
  int j0 = (tile & 63) << 4;

  v8f cr0 = {}, cz0 = {}, chn0 = {}, cin0 = {};
  v8f cr1 = {}, cz1 = {}, chn1 = {}, cin1 = {};
  const _Float16* Ah0 = hprev16 + (size_t)m0 * H_;
  const _Float16* Ah1 = hprev16 + (size_t)(m0 + 16) * H_;
  const _Float16* Ax0 = x16 + (size_t)m0 * H_;
  const _Float16* Ax1 = x16 + (size_t)(m0 + 16) * H_;
  const _Float16* WhR = Whh + (size_t)(j0) * H_;
  const _Float16* WhZ = Whh + (size_t)(H_ + j0) * H_;
  const _Float16* WhN = Whh + (size_t)(2 * H_ + j0) * H_;
  const _Float16* WiR = Wih + (size_t)(j0) * H_;
  const _Float16* WiZ = Wih + (size_t)(H_ + j0) * H_;
  const _Float16* WiN = Wih + (size_t)(2 * H_ + j0) * H_;
  for (int k0 = 0; k0 < H_; k0 += 32) {
    v16h ah0 = load_tile16x32(Ah0, H_, k0);
    v16h ah1 = load_tile16x32(Ah1, H_, k0);
    v16h ax0 = load_tile16x32(Ax0, H_, k0);
    v16h ax1 = load_tile16x32(Ax1, H_, k0);
    v16h w;
    w = load_tile16x32(WhR, H_, k0);
    cr0  = wmma_f16(ah0, w, cr0);   cr1  = wmma_f16(ah1, w, cr1);
    w = load_tile16x32(WiR, H_, k0);
    cr0  = wmma_f16(ax0, w, cr0);   cr1  = wmma_f16(ax1, w, cr1);
    w = load_tile16x32(WhZ, H_, k0);
    cz0  = wmma_f16(ah0, w, cz0);   cz1  = wmma_f16(ah1, w, cz1);
    w = load_tile16x32(WiZ, H_, k0);
    cz0  = wmma_f16(ax0, w, cz0);   cz1  = wmma_f16(ax1, w, cz1);
    w = load_tile16x32(WhN, H_, k0);
    chn0 = wmma_f16(ah0, w, chn0);  chn1 = wmma_f16(ah1, w, chn1);
    w = load_tile16x32(WiN, H_, k0);
    cin0 = wmma_f16(ax0, w, cin0);  cin1 = wmma_f16(ax1, w, cin1);
  }

  int j = j0 + ((threadIdx.x & 31) & 15);
  float bihr = bih[j], bihz = bih[H_ + j], bihn = bih[2 * H_ + j];
  float bhhr = bhh[j], bhhz = bhh[H_ + j], bhhn = bhh[2 * H_ + j];
  gru1_update(cr0, cz0, chn0, cin0, m0, j, bihr, bihz, bihn, bhhr, bhhz, bhhn,
              hprev32, hnew32, hnew16);
  gru1_update(cr1, cz1, chn1, cin1, m0 + 16, j, bihr, bihz, bihn,
              bhhr, bhhz, bhhn, hprev32, hnew32, hnew16);
}

// ---------------------------- output projection ----------------------------
// logits[b, t, v] = h1[b,:] . Wout[v,:] + bout[v]; one block per batch row,
// one wave per vocab entry (looped), wave32 shuffle reduction.
__global__ __launch_bounds__(256) void out_proj_k(
    const float* __restrict__ h32, const float* __restrict__ Wout,
    const float* __restrict__ bout, float* __restrict__ logits, int t) {
  int b    = blockIdx.x;
  int wave = threadIdx.x >> 5;
  int lane = threadIdx.x & 31;
  const float* hr = h32 + (size_t)b * H_;
  for (int v = wave; v < V_; v += 8) {
    const float* wr = Wout + (size_t)v * H_;
    float s = 0.f;
    for (int h = lane; h < H_; h += 32) s += hr[h] * wr[h];
    for (int off = 16; off; off >>= 1) s += __shfl_xor(s, off, 32);
    if (lane == 0)
      logits[(size_t)b * (L_ * V_) + (size_t)t * V_ + v] = s + bout[v];
  }
}

// ------------------------------- launcher ----------------------------------

extern "C" void kernel_launch(void* const* d_in, const int* in_sizes, int n_in,
                              void* d_out, int out_size, void* d_ws,
                              size_t ws_size, hipStream_t stream) {
  const float* z      = (const float*)d_in[0];
  const int*   tseq   = (const int*)d_in[1];
  const float* W_l2h  = (const float*)d_in[2];
  const float* b_l2h  = (const float*)d_in[3];
  const float* Wih0   = (const float*)d_in[4];
  const float* Whh0   = (const float*)d_in[5];
  const float* bih0   = (const float*)d_in[6];
  const float* bhh0   = (const float*)d_in[7];
  const float* Wih1   = (const float*)d_in[8];
  const float* Whh1   = (const float*)d_in[9];
  const float* bih1   = (const float*)d_in[10];
  const float* bhh1   = (const float*)d_in[11];
  const float* Wout   = (const float*)d_in[12];
  const float* bout   = (const float*)d_in[13];
  float* logits = (float*)d_out;

  // Workspace carve-up (all offsets 256B-aligned; total ~30 MB).
  char* ws = (char*)d_ws;
  const size_t WSZ = (size_t)3 * H_ * H_ * sizeof(_Float16);  // 6 MB each
  _Float16* Whh0h = (_Float16*)(ws + 0 * WSZ);
  _Float16* Whh1h = (_Float16*)(ws + 1 * WSZ);
  _Float16* Wih1h = (_Float16*)(ws + 2 * WSZ);
  char* p = ws + 3 * WSZ;
  const size_t HF32 = (size_t)B_ * H_ * sizeof(float);     // 2 MB
  const size_t HF16 = (size_t)B_ * H_ * sizeof(_Float16);  // 1 MB
  float*    h0f32[2] = {(float*)(p), (float*)(p + HF32)};              p += 2 * HF32;
  float*    h1f32[2] = {(float*)(p), (float*)(p + HF32)};              p += 2 * HF32;
  _Float16* h0f16[2] = {(_Float16*)(p), (_Float16*)(p + HF16)};        p += 2 * HF16;
  _Float16* h1f16[2] = {(_Float16*)(p), (_Float16*)(p + HF16)};        p += 2 * HF16;

  // Weights -> f16 (L2-resident working set).
  const int NW = 3 * H_ * H_;
  f32_to_f16_k<<<(NW + 255) / 256, 256, 0, stream>>>(Whh0, Whh0h, NW);
  f32_to_f16_k<<<(NW + 255) / 256, 256, 0, stream>>>(Whh1, Whh1h, NW);
  f32_to_f16_k<<<(NW + 255) / 256, 256, 0, stream>>>(Wih1, Wih1h, NW);

  // Initial hidden state for both layers.
  init_h_k<<<(B_ * 2 * H_) / 256, 256, 0, stream>>>(
      z, W_l2h, b_l2h, h0f32[0], h0f16[0], h1f32[0], h1f16[0]);

  // Sequential decode: 3 launches per step, ping-pong state buffers.
  for (int t = 0; t < L_; ++t) {
    int cur = t & 1, nxt = cur ^ 1;
    gru0_step_k<<<128, 256, 0, stream>>>(
        h0f16[cur], h0f32[cur], h0f32[nxt], h0f16[nxt],
        Whh0h, Wih0, tseq, bih0, bhh0, t);
    gru1_step_k<<<128, 256, 0, stream>>>(
        h1f16[cur], h1f32[cur], h0f16[nxt],
        h1f32[nxt], h1f16[nxt], Whh1h, Wih1h, bih1, bhh1);
    out_proj_k<<<B_, 256, 0, stream>>>(h1f32[nxt], Wout, bout, logits, t);
  }
}
